// EWGNNConv_84490596647561
// MI455X (gfx1250) — matrix operand order repaired and checked by reading, not compiled, not run on hardware
//
#include <hip/hip_runtime.h>
#include <hip/hip_bf16.h>

typedef __bf16 v16bf __attribute__((ext_vector_type(16)));
typedef float  v8f   __attribute__((ext_vector_type(8)));

union Frag { unsigned int u[8]; v16bf v; };

#define CH 128   // IN == H*D == 128
#define HH 4
#define DD 32

// ---------- bf16 helpers (round-to-nearest-even) ----------
__device__ __forceinline__ unsigned int f2bf_raw(float f) {
    union { float f; unsigned int u; } x; x.f = f;
    return (x.u + 0x7FFFu + ((x.u >> 16) & 1u)) >> 16;
}
__device__ __forceinline__ unsigned int pack2(float a, float b) {
    return (f2bf_raw(a) & 0xFFFFu) | (f2bf_raw(b) << 16);
}
__device__ __forceinline__ float bf2f(unsigned short h) {
    union { unsigned int u; float f; } x; x.u = ((unsigned int)h) << 16;
    return x.f;
}

// ---------- LDS loaders ----------
// Weights W[k][n] (128x128 row-major) -> packed pairs along K: sW[kk*130 + n]
__device__ __forceinline__ void load_w(const float* __restrict__ W,
                                       unsigned int* sW, int tid) {
    for (int idx = tid; idx < 64 * CH; idx += 256) {
        int kk = idx >> 7, c = idx & 127;
        sW[kk * 130 + c] = pack2(W[(2 * kk) * CH + c], W[(2 * kk + 1) * CH + c]);
    }
}
// Input tile rows [base, base+128) of X[N][128] -> sX[r*65 + kk] packed K-pairs
__device__ __forceinline__ void load_x(const float* __restrict__ X,
                                       unsigned int* sX, int base, int nN, int tid) {
    for (int idx = tid; idx < 128 * 64; idx += 256) {
        int r = idx >> 6, kk = idx & 63;
        int grow = base + r;
        float f0 = 0.f, f1 = 0.f;
        if (grow < nN) {
            const float* p = X + (size_t)grow * CH + 2 * kk;
            f0 = p[0]; f1 = p[1];
        }
        sX[r * 65 + kk] = pack2(f0, f1);
    }
}

// ---------- WMMA fragment loads (CDNA5 16-bit layouts) ----------
// A 16x32: lanes 0-15 row m, K {0..7,16..23}; lanes 16-31 row m, K {8..15,24..31}
__device__ __forceinline__ void load_afrag(const unsigned int* sX, int mbase, int kt,
                                           int lane, unsigned int* u) {
    int m = lane & 15, half = lane >> 4;
    const unsigned int* row = sX + (mbase + m) * 65;
    int k0 = kt * 16;          // packed-uint base (32 K values = 16 uints)
    int off = half * 4;
#pragma unroll
    for (int v = 0; v < 4; ++v) u[v] = row[k0 + off + v];
#pragma unroll
    for (int v = 0; v < 4; ++v) u[4 + v] = row[k0 + 8 + off + v];
}
// B 32x16: lane holds column n = lane&15; lanes 0-15 K {0..7,16..23}, lanes 16-31 K {8..15,24..31}
__device__ __forceinline__ void load_bfrag(const unsigned int* sW, int ct, int kt,
                                           int lane, unsigned int* u) {
    int n = ct * 16 + (lane & 15), half = lane >> 4;
    int k0 = kt * 16 + half * 4;
#pragma unroll
    for (int v = 0; v < 8; ++v) u[v] = sW[(k0 + v + (v >= 4 ? 4 : 0)) * 130 + n];
}

// Two column tiles (ct0, ct0+1) share the same A fragment: one A load, two
// independent WMMA chains -> fills the WMMA->WMMA hazard window.
__device__ __forceinline__ void mm_tile_pair(const unsigned int* sX,
                                             const unsigned int* sW,
                                             int mbase, int ct0, int lane,
                                             v8f& acc0, v8f& acc1) {
    acc0 = (v8f){0.f, 0.f, 0.f, 0.f, 0.f, 0.f, 0.f, 0.f};
    acc1 = (v8f){0.f, 0.f, 0.f, 0.f, 0.f, 0.f, 0.f, 0.f};
#pragma unroll
    for (int kt = 0; kt < 4; ++kt) {
        Frag a, b0, b1;
        load_afrag(sX, mbase, kt, lane, a.u);
        load_bfrag(sW, ct0,     kt, lane, b0.u);
        load_bfrag(sW, ct0 + 1, kt, lane, b1.u);
        acc0 = __builtin_amdgcn_wmma_f32_16x16x32_bf16(
            false, a.v, false, b0.v, (short)0, acc0, false, false);
        acc1 = __builtin_amdgcn_wmma_f32_16x16x32_bf16(
            false, a.v, false, b1.v, (short)0, acc1, false, false);
    }
}

// ---------- Kernel 0: zero accumulators ----------
__global__ __launch_bounds__(256) void init_kernel(float* p, long long count) {
    long long t = (long long)blockIdx.x * 256 + threadIdx.x;
    if (t < count) p[t] = 0.f;
}

// ---------- Kernel 1: fused q/k/v GEMM + reductions ----------
__global__ __launch_bounds__(256) void qkv_kernel(
    const float* __restrict__ xq, const float* __restrict__ xs,
    const float* __restrict__ Wq, const float* __restrict__ bq,
    const float* __restrict__ Wk, const float* __restrict__ bk,
    const float* __restrict__ Wv, const float* __restrict__ bv,
    float* __restrict__ q_out, float* __restrict__ vbar,
    float* __restrict__ g_ksum, float* __restrict__ g_vsum,
    float* __restrict__ g_kv, float* __restrict__ g_ssq, int nN) {
    __shared__ unsigned int  sW[64 * 130];     // 33,280 B  (current weight, bf16 pairs)
    __shared__ unsigned int  sX[128 * 65];     // 33,280 B  (current input tile, bf16 pairs)
    __shared__ unsigned short sK[128 * 130];   // 33,280 B  (k tile, bf16)
    __shared__ unsigned short sV[128 * 130];   // 33,280 B  (v tile, bf16)
    __shared__ float sKsum[CH], sVsum[CH], sSsq[2];

    const int tid  = threadIdx.x;
    const int lane = tid & 31;
    const int mbase = (tid >> 5) * 16;         // 8 waves x 16 rows
    const int base = blockIdx.x * 128;

    if (tid < CH) { sKsum[tid] = 0.f; sVsum[tid] = 0.f; }
    if (tid < 2)  sSsq[tid] = 0.f;
    float ssq_q = 0.f, ssq_k = 0.f;

    // ---- phase Q ----
    load_w(Wq, sW, tid);
    load_x(xq, sX, base, nN, tid);
    // warm L2/WGP$ for the source tile consumed in phases K/V (global_prefetch_b8)
    for (int r = tid >> 1; r < 128; r += 128) {
        int grow = base + r;
        if (grow < nN)
            __builtin_prefetch(xs + (size_t)grow * CH + (tid & 1) * 64, 0, 1);
    }
    __syncthreads();
    for (int p = 0; p < 4; ++p) {
        v8f acc[2];
        mm_tile_pair(sX, sW, mbase, 2 * p, lane, acc[0], acc[1]);
#pragma unroll
        for (int t = 0; t < 2; ++t) {
            int col = (2 * p + t) * 16 + (lane & 15);
            float b = bq[col];
            int rbase = mbase + ((lane >> 4) << 3);
#pragma unroll
            for (int r = 0; r < 8; ++r) {
                int grow = base + rbase + r;
                if (grow < nN) {
                    float c = acc[t][r] + b;
                    ssq_q += c * c;
                    q_out[(size_t)grow * CH + col] = c;
                }
            }
        }
    }
    __syncthreads();

    // ---- phase K ----
    load_w(Wk, sW, tid);
    load_x(xs, sX, base, nN, tid);
    __syncthreads();
    for (int p = 0; p < 4; ++p) {
        v8f acc[2];
        mm_tile_pair(sX, sW, mbase, 2 * p, lane, acc[0], acc[1]);
#pragma unroll
        for (int t = 0; t < 2; ++t) {
            int col = (2 * p + t) * 16 + (lane & 15);
            float b = bk[col];
            int rbase = mbase + ((lane >> 4) << 3);
            float colsum = 0.f;
#pragma unroll
            for (int r = 0; r < 8; ++r) {
                int grow = base + rbase + r;
                float c = (grow < nN) ? (acc[t][r] + b) : 0.f;
                ssq_k += c * c;
                colsum += c;
                sK[(rbase + r) * 130 + col] = (unsigned short)f2bf_raw(c);
            }
            atomicAdd(&sKsum[col], colsum);
        }
    }
    __syncthreads();

    // ---- phase V (sX still holds source tile) ----
    load_w(Wv, sW, tid);
    __syncthreads();
    for (int p = 0; p < 4; ++p) {
        v8f acc[2];
        mm_tile_pair(sX, sW, mbase, 2 * p, lane, acc[0], acc[1]);
#pragma unroll
        for (int t = 0; t < 2; ++t) {
            int col = (2 * p + t) * 16 + (lane & 15);
            float b = bv[col];
            int rbase = mbase + ((lane >> 4) << 3);
            float colsum = 0.f;
#pragma unroll
            for (int r = 0; r < 8; ++r) {
                int grow = base + rbase + r;
                float c = (grow < nN) ? (acc[t][r] + b) : 0.f;
                colsum += c;
                sV[(rbase + r) * 130 + col] = (unsigned short)f2bf_raw(c);
            }
            atomicAdd(&sVsum[col], colsum);
        }
    }
    atomicAdd(&sSsq[0], ssq_q);
    atomicAdd(&sSsq[1], ssq_k);
    __syncthreads();

    // ---- vbar = mean over heads of v ----
    for (int idx = tid; idx < 128 * DD; idx += 256) {
        int r = idx >> 5, d = idx & 31;
        int grow = base + r;
        if (grow < nN) {
            const unsigned short* vr = &sV[r * 130];
            float s = bf2f(vr[d]) + bf2f(vr[32 + d]) + bf2f(vr[64 + d]) + bf2f(vr[96 + d]);
            vbar[(size_t)grow * DD + d] = 0.25f * s;
        }
    }
    // ---- kv[h][m][d] += sum_rows k*v (per-block partial -> global atomics) ----
    for (int cell = tid; cell < HH * DD * DD; cell += 256) {
        int h = cell >> 10, m = (cell >> 5) & 31, d = cell & 31;
        const unsigned short* kp = &sK[h * 32 + m];
        const unsigned short* vp = &sV[h * 32 + d];
        float a = 0.f;
#pragma unroll 4
        for (int r = 0; r < 128; ++r)
            a += bf2f(kp[r * 130]) * bf2f(vp[r * 130]);
        atomicAdd(&g_kv[cell], a);
    }
    // ---- flush block partials ----
    if (tid < CH) {
        atomicAdd(&g_ksum[tid], sKsum[tid]);
        atomicAdd(&g_vsum[tid], sVsum[tid]);
    }
    if (tid == 0) atomicAdd(&g_ssq[0], sSsq[0]);
    if (tid == 1) atomicAdd(&g_ssq[1], sSsq[1]);
}

// ---------- Kernel 2: in-degree by destination (col) ----------
__global__ __launch_bounds__(256) void deg_kernel(const int* __restrict__ ei,
                                                  float* __restrict__ deg, int E) {
    int t = blockIdx.x * 256 + threadIdx.x;
    if (t < E) atomicAdd(&deg[ei[(size_t)E + t]], 1.0f);
}

// ---------- Kernel 3: edge scatter on vbar (32 lanes per edge) ----------
__global__ __launch_bounds__(256) void gc_kernel(const int* __restrict__ ei,
                                                 const float* __restrict__ deg,
                                                 const float* __restrict__ vbar,
                                                 float* __restrict__ gc, int E) {
    size_t t = (size_t)blockIdx.x * 256 + threadIdx.x;
    size_t e = t >> 5;
    int d = (int)(t & 31);
    if (e < (size_t)E) {
        int row = ei[e];
        int col = ei[(size_t)E + e];
        float dr = deg[row], dc = deg[col];
        float w = (dr > 0.f && dc > 0.f) ? rsqrtf(dr) * rsqrtf(dc) : 0.f;
        if (w != 0.f)
            atomicAdd(&gc[(size_t)col * DD + d], w * vbar[(size_t)row * DD + d]);
    }
}

// ---------- Kernel 4: attention finalize + combine ----------
__global__ __launch_bounds__(256) void final_kernel(
    const float* __restrict__ q_buf, const float* __restrict__ g_ksum,
    const float* __restrict__ g_vsum, const float* __restrict__ g_kv,
    const float* __restrict__ g_ssq, const float* __restrict__ gc,
    float* __restrict__ out, int nN) {
    __shared__ float sKV[HH * DD * DD];
    __shared__ float sKs[CH], sVs[CH];
    __shared__ float sQ[8 * CH];
    __shared__ float sScale;

    const int tid = threadIdx.x;
    for (int i = tid; i < HH * DD * DD; i += 256) sKV[i] = g_kv[i];
    if (tid < CH) { sKs[tid] = g_ksum[tid]; sVs[tid] = g_vsum[tid]; }
    if (tid == 0) {
        float nq = sqrtf(g_ssq[0]), nk = sqrtf(g_ssq[1]);
        sScale = 1.0f / (nq * nk);
    }
    for (int j = 0; j < 4; ++j) {
        int idx = tid + j * 256;           // 8 rows x 128 cols
        int r = idx >> 7, c = idx & 127;
        int gn = blockIdx.x * 8 + r;
        sQ[idx] = (gn < nN) ? q_buf[(size_t)gn * CH + c] : 0.f;
    }
    __syncthreads();

    const int ln = tid >> 5;
    const int n = blockIdx.x * 8 + ln;
    const int d = tid & 31;
    if (n < nN) {
        const float scale = sScale;
        const float Nf = (float)nN;
        const float* q = &sQ[ln * CH];
        float acc = 0.f;
#pragma unroll
        for (int h = 0; h < HH; ++h) {
            const float* kvh = &sKV[h * DD * DD];
            const float* qh = q + h * DD;
            float t = 0.f, s = 0.f;
#pragma unroll 4
            for (int m = 0; m < DD; ++m) {
                float qm = qh[m];
                t += qm * kvh[m * DD + d];
                s += qm * sKs[h * DD + m];
            }
            float num = scale * t + sVs[h * DD + d];
            float den = scale * s + Nf;
            acc += num / den;
        }
        out[(size_t)n * DD + d] = 0.25f * acc + gc[(size_t)n * DD + d];
    }
}

extern "C" void kernel_launch(void* const* d_in, const int* in_sizes, int n_in,
                              void* d_out, int out_size, void* d_ws, size_t ws_size,
                              hipStream_t stream) {
    const float* xq = (const float*)d_in[0];
    const float* xs = (const float*)d_in[1];
    const int*   ei = (const int*)d_in[2];
    const float* Wq = (const float*)d_in[3];
    const float* bq = (const float*)d_in[4];
    const float* Wk = (const float*)d_in[5];
    const float* bk = (const float*)d_in[6];
    const float* Wv = (const float*)d_in[7];
    const float* bv = (const float*)d_in[8];
    const int N = in_sizes[0] / CH;
    const int E = in_sizes[2] / 2;

    float* ws    = (float*)d_ws;
    float* q_buf = ws;                                   // N*128
    float* vbar  = ws + (size_t)N * CH;                  // N*32
    float* stats = vbar + (size_t)N * DD;
    float* ssq   = stats;                                // 2
    float* ksum  = stats + 2;                            // 128
    float* vsum  = stats + 130;                          // 128
    float* kv    = stats + 258;                          // 4096
    float* deg   = stats + 4354;                         // N
    float* gc    = deg + N;                              // N*32

    long long zcount = 4354LL + 33LL * N;                // stats + deg + gc
    init_kernel<<<(unsigned)((zcount + 255) / 256), 256, 0, stream>>>(stats, zcount);

    qkv_kernel<<<(N + 127) / 128, 256, 0, stream>>>(
        xq, xs, Wq, bq, Wk, bk, Wv, bv, q_buf, vbar, ksum, vsum, kv, ssq, N);

    deg_kernel<<<(E + 255) / 256, 256, 0, stream>>>(ei, deg, E);

    size_t gthreads = (size_t)E * 32;
    gc_kernel<<<(unsigned)((gthreads + 255) / 256), 256, 0, stream>>>(ei, deg, vbar, gc, E);

    final_kernel<<<(N + 7) / 8, 256, 0, stream>>>(q_buf, ksum, vsum, kv, ssq, gc,
                                                  (float*)d_out, N);
}
